// ExpanderGINLayer_81149112091150
// MI455X (gfx1250) — matrix operand order repaired and checked by reading, not compiled, not run on hardware
//
#include <hip/hip_runtime.h>

// ---------------------------------------------------------------------------
// ExpanderGIN layer for MI455X (gfx1250, wave32, WMMA).
//   neigh = segment_sum(h[src] -> dst); x = (1+eps)h + neigh
//   x = relu(x@W1+b1)@W2+b2; x *= snorm; batchnorm; relu; + h
// Memory-bound pipeline. GEMMs: v_wmma_f32_16x16x32_bf16 with weights
// pre-packed into per-lane B-fragment order (ISA 05_wmma §7.12.2) and staged
// block-wide into LDS via global_load_async_to_lds_b128 (ISA 08_async_tensor)
// + s_wait_asynccnt. B fragments are preloaded per K-chunk into a register
// array so ds_load latency is hidden under the WMMA burst (staggered
// s_wait_dscnt instead of load->wait->wmma serialization).
// ---------------------------------------------------------------------------

#define N_NODES 100000
#define D 128
#define N_EDGES 1600000
#define N_TILES (N_NODES / 16)      // 6250 exact
#define TOT (N_NODES * D)           // 12.8M
#define BN_EPS 1e-5f

typedef __attribute__((ext_vector_type(16))) __bf16           v16bf;
typedef __attribute__((ext_vector_type(8)))  float            v8f;
typedef __attribute__((ext_vector_type(16))) unsigned short   v16us;
typedef __attribute__((ext_vector_type(8)))  unsigned short   v8us;

union FragU { v16us us; v16bf bf; };

static __device__ __forceinline__ unsigned short f32_to_bf16(float f) {
  union { float f; unsigned int u; } v; v.f = f;
  unsigned int u = v.u;
  unsigned int rnd = 0x7FFFu + ((u >> 16) & 1u);   // round-to-nearest-even
  return (unsigned short)((u + rnd) >> 16);
}

// Async-copy one 32 KB packed-W matrix into LDS, block-cooperative.
// 128 threads x 16 iters x 16 B = 32768 B. Per ISA 08_async_tensor §4:
// each lane transfers 16 B; tracked with ASYNCcnt.
static __device__ __forceinline__ void stage_W_to_lds(unsigned short* sW,
                                                      const unsigned short* Wp) {
  unsigned t = (unsigned)threadIdx.x;                 // 0..127
  unsigned lds_base = (unsigned)(unsigned long long)(void*)sW;  // addr[31:0] = LDS offset
  unsigned long long g = (unsigned long long)Wp;
#pragma unroll
  for (int i = 0; i < 16; ++i) {
    unsigned boff = t * 16u + (unsigned)i * 2048u;    // 128 thr * 16 B = 2 KB / iter
    unsigned lds = lds_base + boff;
    unsigned long long ga = g + boff;
    asm volatile("global_load_async_to_lds_b128 %0, %1, off"
                 :: "v"(lds), "v"(ga) : "memory");
  }
  asm volatile("s_wait_asynccnt 0" ::: "memory");
  __syncthreads();
}

// --- 1. pack W (f32 row-major [k][n], 128x128) into bf16 WMMA B-fragments ---
// frag = (kk*8 + nt), lane 0..31, element e 0..15:
//   k = kk*32 + 8*(lane>>4) + (e&7) + 16*(e>>3) ; n = nt*16 + (lane&15)
__global__ void k_pack_weights(const float* __restrict__ W1,
                               const float* __restrict__ W2,
                               unsigned short* __restrict__ W1p,
                               unsigned short* __restrict__ W2p) {
  int i = blockIdx.x * 256 + threadIdx.x;
  if (i >= 2 * 16384) return;
  const float* W = (i < 16384) ? W1 : W2;
  unsigned short* Wp = (i < 16384) ? W1p : W2p;
  int j    = i & 16383;
  int e    = j & 15;
  int lane = (j >> 4) & 31;
  int nt   = (j >> 9) & 7;
  int kk   = (j >> 12) & 3;
  int n = nt * 16 + (lane & 15);
  int k = kk * 32 + (lane >> 4) * 8 + (e & 7) + (e >> 3) * 16;
  Wp[j] = f32_to_bf16(W[k * D + n]);
}

// --- 2. xacc = (1+eps) * h  (float4 streaming) -----------------------------
__global__ void k_init_acc(const float* __restrict__ h,
                           const float* __restrict__ eps,
                           float* __restrict__ xacc) {
  int i = blockIdx.x * 256 + threadIdx.x;     // over TOT/4 float4s
  if (i >= TOT / 4) return;
  float s = 1.0f + eps[0];
  float4 v = ((const float4*)h)[i];
  float4 r; r.x = v.x * s; r.y = v.y * s; r.z = v.z * s; r.w = v.w * s;
  ((float4*)xacc)[i] = r;
}

// --- 3. edge scatter: wave32 per edge, lane = 4 contiguous floats ----------
// h (51 MB) and xacc (51 MB) both fit in the 192 MB L2 -> L2-atomic bound.
__global__ void k_edge_scatter(const float* __restrict__ h,
                               const int* __restrict__ esrc,
                               const int* __restrict__ edst,
                               float* __restrict__ xacc) {
  int e = blockIdx.x * 8 + (threadIdx.x >> 5);
  if (e >= N_EDGES) return;
  int lane = threadIdx.x & 31;
  int src = esrc[e];
  int dst = edst[e];
  float4 v = *(const float4*)(h + (size_t)src * D + lane * 4);
  float* p = xacc + (size_t)dst * D + lane * 4;
  atomicAdd(p + 0, v.x);
  atomicAdd(p + 1, v.y);
  atomicAdd(p + 2, v.z);
  atomicAdd(p + 3, v.w);
}

// --- 4. zero BN stats ------------------------------------------------------
__global__ void k_zero_stats(float* __restrict__ sums, float* __restrict__ sumsq) {
  int t = threadIdx.x;
  if (t < D) { sums[t] = 0.0f; sumsq[t] = 0.0f; }
}

// --- 5. GEMM1: y1 = relu(xacc @ W1 + b1), f32 in -> bf16 out ---------------
// One wave per 16-row M tile; 8 accumulators = full N=128; K in 4x32.
// A-fragment per ISA 16-bit A layout: lane (m = lane&15, h = lane>>4):
//   elements 0..7  -> K = kk*32 + 8h + 0..7      (contiguous)
//   elements 8..15 -> K = kk*32 + 16 + 8h + 0..7 (contiguous)
__global__ void k_gemm1(const float* __restrict__ xacc,
                        const unsigned short* __restrict__ W1p,
                        const float* __restrict__ b1,
                        unsigned short* __restrict__ y1) {
  __shared__ __align__(128) unsigned short sW[16384];   // 32 KB packed W1
  stage_W_to_lds(sW, W1p);                              // all 128 threads

  const int lane = threadIdx.x & 31;
  const int wave = threadIdx.x >> 5;
  const int tile = blockIdx.x * 4 + wave;
  if (tile >= N_TILES) return;                 // wave-uniform: EXEC stays all-1
  const int m0 = tile * 16;
  const int mrow = m0 + (lane & 15);
  const int hh = lane >> 4;

  // Hoist all A global loads (8 x float4x2 = whole row's worth for this lane):
  // max memory-level parallelism, one staggered wait instead of per-chunk stalls.
  const float* abase = xacc + (size_t)mrow * D + hh * 8;
  float4 araw[4][4];                           // [kk][run0_lo,run0_hi,run1_lo,run1_hi]
#pragma unroll
  for (int kk = 0; kk < 4; ++kk) {
    const float* arow = abase + kk * 32;
    araw[kk][0] = *(const float4*)(arow + 0);
    araw[kk][1] = *(const float4*)(arow + 4);
    araw[kk][2] = *(const float4*)(arow + 16);
    araw[kk][3] = *(const float4*)(arow + 20);
  }

  v8f acc[8];
#pragma unroll
  for (int i = 0; i < 8; ++i) acc[i] = (v8f)(0.0f);

#pragma unroll
  for (int kk = 0; kk < 4; ++kk) {
    // Preload all 8 B fragments of this K-chunk into distinct registers so the
    // 16 ds_load_b128 issue as a batch and drain under the WMMA burst.
    FragU b[8];
#pragma unroll
    for (int nt = 0; nt < 8; ++nt)
      b[nt].us = *(const v16us*)(sW + (((kk * 8 + nt) * 32 + lane) << 4));

    FragU a;
#pragma unroll
    for (int q = 0; q < 4; ++q) {
      const float4 v = araw[kk][q];
      a.us[q * 4 + 0] = f32_to_bf16(v.x);
      a.us[q * 4 + 1] = f32_to_bf16(v.y);
      a.us[q * 4 + 2] = f32_to_bf16(v.z);
      a.us[q * 4 + 3] = f32_to_bf16(v.w);
    }

#pragma unroll
    for (int nt = 0; nt < 8; ++nt)
      acc[nt] = __builtin_amdgcn_wmma_f32_16x16x32_bf16(
          false, a.bf, false, b[nt].bf, (short)0, acc[nt], false, false);
  }

  // Epilogue: C layout -> lane holds n = lane&15; VGPR r -> m = m0 + 8*hh + r.
  const int n0 = lane & 15;
#pragma unroll
  for (int nt = 0; nt < 8; ++nt) {
    int n = nt * 16 + n0;
    float bias = b1[n];
#pragma unroll
    for (int r = 0; r < 8; ++r) {
      int m = m0 + hh * 8 + r;
      float v = acc[nt][r] + bias;
      v = v > 0.0f ? v : 0.0f;
      y1[(size_t)m * D + n] = f32_to_bf16(v);
    }
  }
}

// --- 6. GEMM2: x2 = (y1 @ W2 + b2) * snorm, bf16 in -> f32 out -------------
__global__ void k_gemm2(const unsigned short* __restrict__ y1,
                        const unsigned short* __restrict__ W2p,
                        const float* __restrict__ b2,
                        const float* __restrict__ snorm,
                        float* __restrict__ x2) {
  __shared__ __align__(128) unsigned short sW[16384];   // 32 KB packed W2
  stage_W_to_lds(sW, W2p);

  const int lane = threadIdx.x & 31;
  const int wave = threadIdx.x >> 5;
  const int tile = blockIdx.x * 4 + wave;
  if (tile >= N_TILES) return;
  const int m0 = tile * 16;
  const int mrow = m0 + (lane & 15);
  const int hh = lane >> 4;

  // Hoist all A global loads (bf16 rows, 8 x 16B).
  const unsigned short* abase = y1 + (size_t)mrow * D + hh * 8;
  v8us araw[4][2];
#pragma unroll
  for (int kk = 0; kk < 4; ++kk) {
    araw[kk][0] = *(const v8us*)(abase + kk * 32);
    araw[kk][1] = *(const v8us*)(abase + kk * 32 + 16);
  }

  v8f acc[8];
#pragma unroll
  for (int i = 0; i < 8; ++i) acc[i] = (v8f)(0.0f);

#pragma unroll
  for (int kk = 0; kk < 4; ++kk) {
    FragU b[8];
#pragma unroll
    for (int nt = 0; nt < 8; ++nt)
      b[nt].us = *(const v16us*)(sW + (((kk * 8 + nt) * 32 + lane) << 4));

    FragU a;
#pragma unroll
    for (int e = 0; e < 8; ++e) {
      a.us[e]     = araw[kk][0][e];
      a.us[8 + e] = araw[kk][1][e];
    }

#pragma unroll
    for (int nt = 0; nt < 8; ++nt)
      acc[nt] = __builtin_amdgcn_wmma_f32_16x16x32_bf16(
          false, a.bf, false, b[nt].bf, (short)0, acc[nt], false, false);
  }

  const int n0 = lane & 15;
#pragma unroll
  for (int nt = 0; nt < 8; ++nt) {
    int n = nt * 16 + n0;
    float bias = b2[n];
#pragma unroll
    for (int r = 0; r < 8; ++r) {
      int m = m0 + hh * 8 + r;
      float v = (acc[nt][r] + bias) * snorm[m];
      x2[(size_t)m * D + n] = v;
    }
  }
}

// --- 7. BN partial reduce: thread = feature, block-strided over rows -------
__global__ void k_bn_reduce(const float* __restrict__ x2,
                            float* __restrict__ sums,
                            float* __restrict__ sumsq) {
  int f = threadIdx.x;                 // 128 threads
  float s = 0.0f, q = 0.0f;
  for (int row = blockIdx.x; row < N_NODES; row += gridDim.x) {
    float v = x2[(size_t)row * D + f];
    s += v; q += v * v;
  }
  atomicAdd(&sums[f], s);
  atomicAdd(&sumsq[f], q);
}

// --- 8. finalize: normalize, affine, relu, residual ------------------------
__global__ void k_final(const float* __restrict__ h,
                        const float* __restrict__ x2,
                        const float* __restrict__ sums,
                        const float* __restrict__ sumsq,
                        const float* __restrict__ gamma,
                        const float* __restrict__ beta,
                        float* __restrict__ out) {
  int i = blockIdx.x * 256 + threadIdx.x;
  if (i >= TOT) return;
  int f = i & (D - 1);
  const float invN = 1.0f / (float)N_NODES;
  float mean = sums[f] * invN;
  float var = sumsq[f] * invN - mean * mean;
  float inv = rsqrtf(var + BN_EPS);
  float v = (x2[i] - mean) * inv * gamma[f] + beta[f];
  v = v > 0.0f ? v : 0.0f;
  out[i] = h[i] + v;
}

// ---------------------------------------------------------------------------
extern "C" void kernel_launch(void* const* d_in, const int* in_sizes, int n_in,
                              void* d_out, int out_size, void* d_ws, size_t ws_size,
                              hipStream_t stream) {
  const float* h      = (const float*)d_in[0];
  const float* snorm  = (const float*)d_in[1];
  const int*   esrc   = (const int*)d_in[2];
  const int*   edst   = (const int*)d_in[3];
  const float* eps    = (const float*)d_in[4];
  const float* W1     = (const float*)d_in[5];
  const float* b1     = (const float*)d_in[6];
  const float* W2     = (const float*)d_in[7];
  const float* b2     = (const float*)d_in[8];
  const float* gamma  = (const float*)d_in[9];
  const float* beta   = (const float*)d_in[10];
  float* out = (float*)d_out;

  // Workspace layout (all 256B aligned):
  char* ws = (char*)d_ws;
  float*          xacc = (float*)(ws);                              // 51.2 MB (also x2)
  unsigned short* y1   = (unsigned short*)(ws + 51200000);          // 25.6 MB
  unsigned short* W1p  = (unsigned short*)(ws + 51200000 + 25600000);
  unsigned short* W2p  = (unsigned short*)(ws + 51200000 + 25600000 + 32768);
  float*          sums = (float*)(ws + 51200000 + 25600000 + 65536);
  float*          sumsq= (float*)(ws + 51200000 + 25600000 + 65536 + 512);
  float*          x2   = xacc;   // alias: xacc is dead after GEMM1

  // 1. pack weights to WMMA fragment order (bf16)
  k_pack_weights<<<(2 * 16384 + 255) / 256, 256, 0, stream>>>(W1, W2, W1p, W2p);
  // 2. xacc = (1+eps)*h
  k_init_acc<<<(TOT / 4 + 255) / 256, 256, 0, stream>>>(h, eps, xacc);
  // 3. scatter-add neighbors (wave per edge)
  k_edge_scatter<<<(N_EDGES + 7) / 8, 256, 0, stream>>>(h, esrc, edst, xacc);
  // 4. zero BN stats
  k_zero_stats<<<1, 256, 0, stream>>>(sums, sumsq);
  // 5. GEMM1 (WMMA bf16, W1 staged to LDS via async copy), relu, bf16 out
  k_gemm1<<<(N_TILES + 3) / 4, 128, 0, stream>>>(xacc, W1p, b1, y1);
  // 6. GEMM2 (WMMA bf16, W2 staged to LDS), +bias, *snorm, f32 out (aliases xacc)
  k_gemm2<<<(N_TILES + 3) / 4, 128, 0, stream>>>(y1, W2p, b2, snorm, x2);
  // 7. BN stats
  k_bn_reduce<<<512, 128, 0, stream>>>(x2, sums, sumsq);
  // 8. normalize + relu + residual
  k_final<<<(TOT + 255) / 256, 256, 0, stream>>>(h, x2, sums, sumsq, gamma, beta, out);
}
